// GAT2_88356067214038
// MI455X (gfx1250) — matrix operand order repaired and checked by reading, not compiled, not run on hardware
//
#include <hip/hip_runtime.h>
#include <stdint.h>

typedef __attribute__((ext_vector_type(16))) __bf16 v16bf;
typedef __attribute__((ext_vector_type(8)))  __bf16 v8bf;
typedef __attribute__((ext_vector_type(4)))  __bf16 v4bf;
typedef __attribute__((ext_vector_type(8)))  float  v8f;

#define HEADS 8
#define HID 16
#define CH 128            // HEADS*HID == IN_C
#define NEG_SLOPE 0.2f
#define MINF_KEY 0x007FFFFFu   // ordered-uint encoding of -inf

// ---- init / prep kernels -------------------------------------------------

// Convert x (f32, [N][128]) to bf16 copy; 4 elements/thread, packed stores.
__global__ void k_cvt_x(const float* __restrict__ x, __bf16* __restrict__ xb, int total4) {
  int t = blockIdx.x * blockDim.x + threadIdx.x;
  if (t >= total4) return;
  float4 f = ((const float4*)x)[t];
  v4bf b;
  b[0] = (__bf16)f.x; b[1] = (__bf16)f.y; b[2] = (__bf16)f.z; b[3] = (__bf16)f.w;
  *(v4bf*)(xb + (size_t)t * 4) = b;
}

// Transpose weights to bf16 [col][k] so B fragments are contiguous per lane.
__global__ void k_prep_w(const float* __restrict__ Wl, const float* __restrict__ Wr,
                         __bf16* __restrict__ Wlt, __bf16* __restrict__ Wrt) {
  int t = blockIdx.x * blockDim.x + threadIdx.x;
  if (t >= CH * CH) return;
  int c = t >> 7, k = t & 127;
  Wlt[c * CH + k] = (__bf16)Wl[k * CH + c];
  Wrt[c * CH + k] = (__bf16)Wr[k * CH + c];
}

__global__ void k_init_nh(unsigned* __restrict__ m_ord, float* __restrict__ denom, int n8) {
  int t = blockIdx.x * blockDim.x + threadIdx.x;
  if (t < n8) { m_ord[t] = MINF_KEY; denom[t] = 0.f; }
}

__global__ void k_init_out(float* __restrict__ out, int n) {
  int t = blockIdx.x * blockDim.x + threadIdx.x;
  if (t < n) out[t] = 0.f;
}

// ---- projection GEMM: one wave per 16x16 tile of xl AND xr ---------------
// block = 256 threads = 8 waves; wave w handles column tile w (8*16 = 128 cols)

__global__ void __launch_bounds__(256)
k_proj(const __bf16* __restrict__ xb, const __bf16* __restrict__ Wlt,
       const __bf16* __restrict__ Wrt, float* __restrict__ xl,
       float* __restrict__ xr, int N) {
  const int lane    = threadIdx.x & 31;
  const int colTile = threadIdx.x >> 5;        // 0..7
  const int rowTile = blockIdx.x;
  const int m16     = lane & 15;
  const int hi      = lane >> 4;               // 0 or 1 (lane half)

  int row  = rowTile * 16 + m16;
  int rowL = row < N ? row : N - 1;            // clamp for tail safety
  const int nCol = colTile * 16 + m16;         // B-fragment column / D column

  v8f accL = {0.f,0.f,0.f,0.f,0.f,0.f,0.f,0.f};
  v8f accR = {0.f,0.f,0.f,0.f,0.f,0.f,0.f,0.f};

  const __bf16* arow = xb + (size_t)rowL * CH + hi * 16;
  const __bf16* blp  = Wlt + (size_t)nCol * CH + hi * 16;
  const __bf16* brp  = Wrt + (size_t)nCol * CH + hi * 16;

  #pragma unroll
  for (int kt = 0; kt < 4; ++kt) {
    // A fragment (16x32 bf16): elems 0-15 <-> K=kb+16*hi..kb+16*hi+15
    v8bf a0 = *(const v8bf*)(arow + kt * 32);
    v8bf a1 = *(const v8bf*)(arow + kt * 32 + 8);
    v16bf a = __builtin_shufflevector(a0, a1, 0,1,2,3,4,5,6,7,8,9,10,11,12,13,14,15);
    // B fragments: 16 contiguous bf16 (32B) per lane from pre-transposed weights
    v16bf bl = *(const v16bf*)(blp + kt * 32);
    v16bf br = *(const v16bf*)(brp + kt * 32);

    accL = __builtin_amdgcn_wmma_f32_16x16x32_bf16(false, a, false, bl,
                                                   (short)0, accL, false, false);
    accR = __builtin_amdgcn_wmma_f32_16x16x32_bf16(false, a, false, br,
                                                   (short)0, accR, false, false);
  }

  // D store: VGPR v -> row v + 8*hi, col = lane&15.
  // Tile-uniform bound test: unguarded fast path for full tiles (always for N%16==0).
  float* pl = xl + ((size_t)(rowTile * 16 + hi * 8)) * CH + colTile * 16 + m16;
  float* pr = xr + ((size_t)(rowTile * 16 + hi * 8)) * CH + colTile * 16 + m16;
  if (rowTile * 16 + 16 <= N) {
    #pragma unroll
    for (int v = 0; v < 8; ++v) {
      pl[(size_t)v * CH] = accL[v];
      pr[(size_t)v * CH] = accR[v];
    }
  } else {
    #pragma unroll
    for (int v = 0; v < 8; ++v) {
      if (rowTile * 16 + v + hi * 8 < N) {
        pl[(size_t)v * CH] = accL[v];
        pr[(size_t)v * CH] = accR[v];
      }
    }
  }
}

// ---- per-edge score + segment max (one thread per (edge, head)) ----------

__global__ void __launch_bounds__(256)
k_score(const int* __restrict__ ei, const float* __restrict__ xl,
        const float* __restrict__ xr, const float* __restrict__ att,
        float* __restrict__ score, unsigned* __restrict__ m_ord,
        int E, int Etot) {
  int t = blockIdx.x * blockDim.x + threadIdx.x;
  int e = t >> 3, h = t & 7;
  if (e >= Etot) return;
  int src, dst;
  if (e < E) { src = ei[e]; dst = ei[E + e]; } else { src = dst = e - E; }

  const float* pl = xl + (size_t)src * CH + h * HID;
  const float* pr = xr + (size_t)dst * CH + h * HID;
  const float* pa = att + h * HID;
  float s = 0.f;
  #pragma unroll
  for (int c = 0; c < HID; c += 4) {
    float4 a4 = *(const float4*)(pl + c);
    float4 b4 = *(const float4*)(pr + c);
    float4 w4 = *(const float4*)(pa + c);
    float v;
    v = a4.x + b4.x; s += w4.x * (v > 0.f ? v : NEG_SLOPE * v);
    v = a4.y + b4.y; s += w4.y * (v > 0.f ? v : NEG_SLOPE * v);
    v = a4.z + b4.z; s += w4.z * (v > 0.f ? v : NEG_SLOPE * v);
    v = a4.w + b4.w; s += w4.w * (v > 0.f ? v : NEG_SLOPE * v);
  }
  score[(size_t)e * HEADS + h] = s;
  unsigned u = __builtin_bit_cast(unsigned, s);
  unsigned key = (u & 0x80000000u) ? ~u : (u | 0x80000000u);  // order-preserving
  atomicMax(&m_ord[dst * HEADS + h], key);
}

// ---- exp(score - m) + segment sum ----------------------------------------

__global__ void __launch_bounds__(256)
k_exp(const int* __restrict__ ei, const unsigned* __restrict__ m_ord,
      float* __restrict__ score, float* __restrict__ denom, int E, int Etot) {
  int t = blockIdx.x * blockDim.x + threadIdx.x;
  int e = t >> 3, h = t & 7;
  if (e >= Etot) return;
  int dst = (e < E) ? ei[E + e] : (e - E);
  unsigned key = m_ord[dst * HEADS + h];
  unsigned u = (key & 0x80000000u) ? (key ^ 0x80000000u) : ~key;
  float m = __builtin_bit_cast(float, u);
  size_t o = (size_t)e * HEADS + h;
  float a = __expf(score[o] - m);
  score[o] = a;                                  // overwrite score with alpha
  atomicAdd(&denom[dst * HEADS + h], a);
}

// ---- weighted aggregation: one wave per edge, lane = 4 channels ----------

__global__ void __launch_bounds__(256)
k_agg(const int* __restrict__ ei, const float* __restrict__ xl,
      const float* __restrict__ alpha, const float* __restrict__ denom,
      float* __restrict__ out, int E, int Etot) {
  int lane = threadIdx.x & 31;
  int e = blockIdx.x * 8 + (threadIdx.x >> 5);
  if (e >= Etot) return;
  int src, dst;
  if (e < E) { src = ei[e]; dst = ei[E + e]; } else { src = dst = e - E; }
  int c0 = lane * 4;
  int h  = c0 >> 4;
  float w = alpha[(size_t)e * HEADS + h] / denom[dst * HEADS + h];
  float4 v = *(const float4*)(xl + (size_t)src * CH + c0);
  float* po = out + (size_t)dst * CH + c0;
  atomicAdd(po + 0, w * v.x);
  atomicAdd(po + 1, w * v.y);
  atomicAdd(po + 2, w * v.z);
  atomicAdd(po + 3, w * v.w);
}

// ---- bias + ELU ----------------------------------------------------------

__global__ void k_final(float* __restrict__ out, const float* __restrict__ bias, int total) {
  int t = blockIdx.x * blockDim.x + threadIdx.x;
  if (t >= total) return;
  float v = out[t] + bias[t & (CH - 1)];
  out[t] = v > 0.f ? v : expm1f(v);
}

// --------------------------------------------------------------------------

extern "C" void kernel_launch(void* const* d_in, const int* in_sizes, int n_in,
                              void* d_out, int out_size, void* d_ws, size_t ws_size,
                              hipStream_t stream) {
  (void)n_in; (void)out_size; (void)ws_size;
  const float* x    = (const float*)d_in[0];
  const int*   ei   = (const int*)d_in[1];
  const float* Wl   = (const float*)d_in[2];
  const float* Wr   = (const float*)d_in[3];
  const float* att  = (const float*)d_in[4];
  const float* bias = (const float*)d_in[5];
  float* out = (float*)d_out;

  const int N    = in_sizes[0] / CH;
  const int E    = in_sizes[1] / 2;
  const int Etot = E + N;

  char* ws = (char*)d_ws;
  size_t o = 0;
  __bf16*   Wlt   = (__bf16*)(ws + o);   o += (size_t)CH * CH * 2;
  __bf16*   Wrt   = (__bf16*)(ws + o);   o += (size_t)CH * CH * 2;
  __bf16*   xb    = (__bf16*)(ws + o);   o += (size_t)N * CH * 2;
  float*    xl    = (float*)(ws + o);    o += (size_t)N * CH * 4;
  float*    xr    = (float*)(ws + o);    o += (size_t)N * CH * 4;
  float*    score = (float*)(ws + o);    o += (size_t)Etot * HEADS * 4;
  unsigned* m_ord = (unsigned*)(ws + o); o += (size_t)N * HEADS * 4;
  float*    denom = (float*)(ws + o);    o += (size_t)N * HEADS * 4;

  const int total4 = (N * CH) / 4;
  k_cvt_x<<<(total4 + 255) / 256, 256, 0, stream>>>(x, xb, total4);
  k_prep_w<<<(CH * CH + 255) / 256, 256, 0, stream>>>(Wl, Wr, Wlt, Wrt);
  k_init_nh<<<(N * HEADS + 255) / 256, 256, 0, stream>>>(m_ord, denom, N * HEADS);
  k_init_out<<<(N * CH + 255) / 256, 256, 0, stream>>>(out, N * CH);

  k_proj<<<(N + 15) / 16, 256, 0, stream>>>(xb, Wlt, Wrt, xl, xr, N);

  const int th = Etot * HEADS;
  k_score<<<(th + 255) / 256, 256, 0, stream>>>(ei, xl, xr, att, score, m_ord, E, Etot);
  k_exp<<<(th + 255) / 256, 256, 0, stream>>>(ei, m_ord, score, denom, E, Etot);
  k_agg<<<(Etot + 7) / 8, 256, 0, stream>>>(ei, xl, score, denom, out, E, Etot);
  k_final<<<(N * CH + 255) / 256, 256, 0, stream>>>(out, bias, N * CH);
}
